// NetLight_8813272891729
// MI455X (gfx1250) — matrix operand order repaired and checked by reading, not compiled, not run on hardware
//
#include <hip/hip_runtime.h>
#include <hip/hip_bf16.h>

// ---------------------------------------------------------------------------
// Sparse-ish 3D CNN pipeline as bf16 WMMA implicit GEMMs (gfx1250 / MI455X).
//   dense volume: (27,127,127) -> (14,64,64) -> (7,32,32)
//   - channels padded to multiples of 32: each K-chunk of 32 is one
//     (tap, channel-block) -> v_wmma_f32_16x16x32_bf16
//   - A fragments: two global_load_b128 per chunk (channel-major layout
//     matches the ISA A-fragment: K 8h..8h+7 and 16+8h..16+8h+7 per lane half h)
//   - B fragments: weights repacked fragment-native, staged per jz-slab into
//     LDS by the Tensor Data Mover (tensor_load_to_lds + s_wait_tensorcnt),
//     then read with ds_load_b128 and shared by all 8 waves of the block
//   - 2x2 register tiling: each wave computes 32 voxels x 32 channels
//     (4 accumulators) -> 2 memory instructions per WMMA
// ---------------------------------------------------------------------------

typedef __attribute__((ext_vector_type(16))) __bf16 v16bf;
typedef __attribute__((ext_vector_type(8)))  float  v8f;
typedef __attribute__((ext_vector_type(4)))  unsigned u32x4;
typedef __attribute__((ext_vector_type(8)))  int      i32x8;
typedef __attribute__((ext_vector_type(4)))  int      i32x4;

#define D0 27
#define H0 127
#define W0 127
#define V0 (D0*H0*W0)      // 435483
#define D1 14
#define H1 64
#define W1 64
#define V1 (D1*H1*W1)      // 57344
#define D2 7
#define H2 32
#define W2 32
#define V2 (D2*H2*W2)      // 7168
#define NPTS 100000

union ABFrag { v16bf v; unsigned u[8]; uint4 q[2]; };

// ---------------- helpers ----------------

__global__ void zero_bf16(__bf16* __restrict__ p, long n) {
  long i = (long)blockIdx.x * blockDim.x + threadIdx.x;
  if (i < n) p[i] = (__bf16)0.0f;
}

__global__ void scatter_pts(const float* __restrict__ f, const int* __restrict__ co,
                            __bf16* __restrict__ dense, __bf16* __restrict__ m0) {
  int i = blockIdx.x * blockDim.x + threadIdx.x;
  if (i >= NPTS) return;
  int z = co[i*4+1], y = co[i*4+2], x = co[i*4+3];
  long vox = ((long)z*H0 + y)*W0 + x;
  dense[vox*32+0] = (__bf16)f[i*3+0];
  dense[vox*32+1] = (__bf16)f[i*3+1];
  dense[vox*32+2] = (__bf16)f[i*3+2];
  m0[vox] = (__bf16)1.0f;
}

__global__ void mask_downsample(const __bf16* __restrict__ m0, __bf16* __restrict__ md) {
  int i = blockIdx.x * blockDim.x + threadIdx.x;
  if (i >= V1) return;
  int oz = i / (H1*W1), r = i % (H1*W1), oy = r / W1, ox = r - oy*W1;
  float any = 0.f;
  for (int dz = 0; dz < 3; ++dz)
    for (int dy = 0; dy < 3; ++dy)
      for (int dx = 0; dx < 3; ++dx) {
        int iz = 2*oz + dz - 1, iy = 2*oy + dy - 1, ix = 2*ox + dx - 1;
        if ((unsigned)iz < (unsigned)D0 && (unsigned)iy < (unsigned)H0 &&
            (unsigned)ix < (unsigned)W0)
          any += (float)m0[((long)iz*H0 + iy)*W0 + ix];
      }
  md[i] = (__bf16)(any > 0.f ? 1.f : 0.f);
}

// Repack fp32 weights (taps, Cin, Cout) -> bf16 B fragments, tiled so each
// lane's half-fragment is 16B-contiguous and each (chunk, ntile) is a 1024B
// block (TDM-friendly):
//   dword = (((chunk*(Cout/16) + ntile)*2 + kgroup)*16 + nlane)*8 + v
//   chunk = t*(Cinp/32)+c/32 ; kgroup=((c%32)/8)&1 ; v=((c%32)/16)*4+(c%8)/2 ; par=c&1
__global__ void pack_weights(const float* __restrict__ w, unsigned short* __restrict__ wp,
                             int taps, int Cin, int Cinp, int Cout) {
  int idx = blockIdx.x * blockDim.x + threadIdx.x;
  int total = taps * Cinp * Cout;
  if (idx >= total) return;
  int n  = idx % Cout, kk = idx / Cout;
  int t  = kk / Cinp,  c  = kk - t*Cinp;
  float val = (c < Cin) ? w[((long)t*Cin + c)*Cout + n] : 0.0f;
  __bf16 hv = (__bf16)val;
  unsigned short bits;
  __builtin_memcpy(&bits, &hv, sizeof(bits));
  int ci  = c >> 5;
  int cc  = c & 31;
  int kg  = (cc >> 3) & 1;
  int v   = ((cc >> 4) << 2) | ((cc & 7) >> 1);
  int par = cc & 1;
  long chunk = (long)t * (Cinp >> 5) + ci;
  long dw = ((((chunk * (Cout >> 4)) + (n >> 4)) * 2 + kg) * 16 + (n & 15)) * 8 + v;
  wp[dw * 2 + par] = bits;
}

// ---------------- the WMMA implicit-GEMM conv ----------------
// Block: 256 threads = 8 waves; each wave owns a 32-voxel x 32-channel tile
// (2x2 of 16x16 WMMA tiles). grid.y selects a 32-channel pair of N-tiles.
template<int KS, int CINP, bool TRANS>
__global__ __launch_bounds__(256)
void conv_wmma(const __bf16* __restrict__ in, __bf16* __restrict__ out,
               const unsigned char* __restrict__ wbytes, const __bf16* __restrict__ mask,
               int Din, int Hin, int Win, int Dout, int Hout, int Wout,
               int Cout, int stride, int pad) {
  constexpr int NCHK  = CINP >> 5;           // K-chunks per tap
  constexpr int SLAB  = KS * KS * NCHK;      // K-chunks per jz-slab
  __shared__ __align__(1024) unsigned char smem[SLAB * 2048];

  const int lane  = threadIdx.x & 31;
  const int wid   = threadIdx.x >> 5;
  const int Vout  = Dout * Hout * Wout;
  const int khalf = lane >> 4;               // A: K-half ; B: K-group
  const int lidx  = lane & 15;               // A: M row  ; B/C: N column
  const int mbase = (blockIdx.x * 8 + wid) * 32;

  int mvox0 = mbase + lidx;       if (mvox0 >= Vout) mvox0 = Vout - 1;
  int mvox1 = mbase + 16 + lidx;  if (mvox1 >= Vout) mvox1 = Vout - 1;
  const int HWo = Hout * Wout;
  const int oz0 = mvox0 / HWo; int r0 = mvox0 - oz0*HWo;
  const int oy0 = r0 / Wout;   const int ox0 = r0 - oy0*Wout;
  const int oz1 = mvox1 / HWo; int r1 = mvox1 - oz1*HWo;
  const int oy1 = r1 / Wout;   const int ox1 = r1 - oy1*Wout;

  const unsigned ntiles   = (unsigned)(Cout >> 4);
  const unsigned lds_base = (unsigned)(unsigned long long)(const void*)smem;

  v8f acc00 = {}, acc01 = {}, acc10 = {}, acc11 = {};

  for (int jz = 0; jz < KS; ++jz) {
    __syncthreads();                         // previous slab fully consumed
    if (wid == 0) {
      // TDM: DMA this jz-slab of packed B fragments into LDS.
      // 2D tile: SLAB rows x 2048B, row stride = ntiles*1024B.
      unsigned long long gaddr = (unsigned long long)wbytes +
          ((unsigned long long)(jz * SLAB) * ntiles + (unsigned)(blockIdx.y * 2)) * 1024ull;
      u32x4 g0; i32x8 g1; i32x4 g2 = {0,0,0,0}, g3 = {0,0,0,0};
      i32x8 g4 = {0,0,0,0,0,0,0,0};
      g0[0] = 1u;                                           // count=1 (valid D#)
      g0[1] = lds_base;                                     // lds_addr
      g0[2] = (unsigned)gaddr;                              // global_addr[31:0]
      g0[3] = (unsigned)((gaddr >> 32) & 0x1FFFFFFull) | (2u << 30);  // [56:32]+type=2
      const unsigned stride0 = ntiles * 1024u;
      g1[0] = 0;                                            // no multicast, 1B elems
      g1[1] = (int)((2048u & 0xFFFFu) << 16);               // tensor_dim0.lo
      g1[2] = (int)((2048u >> 16) | (((unsigned)SLAB & 0xFFFFu) << 16)); // dim0.hi|dim1.lo
      g1[3] = (int)((((unsigned)SLAB >> 16) & 0xFFFFu) | (2048u << 16)); // dim1.hi|tile_dim0
      g1[4] = (int)((unsigned)SLAB & 0xFFFFu);              // tile_dim1
      g1[5] = (int)stride0;                                 // tensor_dim0_stride.lo
      g1[6] = 0; g1[7] = 0;
      __builtin_amdgcn_tensor_load_to_lds(g0, g1, g2, g3, g4, 0);
      __builtin_amdgcn_s_wait_tensorcnt(0);
    }
    __syncthreads();                         // slab visible to all waves

    for (int jy = 0; jy < KS; ++jy)
      for (int jx = 0; jx < KS; ++jx) {
        int iz0, iy0, ix0, iz1, iy1, ix1; bool val0, val1;
        if (!TRANS) {
          iz0 = oz0*stride + jz - pad; iy0 = oy0*stride + jy - pad; ix0 = ox0*stride + jx - pad;
          val0 = ((unsigned)iz0 < (unsigned)Din) & ((unsigned)iy0 < (unsigned)Hin) &
                 ((unsigned)ix0 < (unsigned)Win);
          iz1 = oz1*stride + jz - pad; iy1 = oy1*stride + jy - pad; ix1 = ox1*stride + jx - pad;
          val1 = ((unsigned)iz1 < (unsigned)Din) & ((unsigned)iy1 < (unsigned)Hin) &
                 ((unsigned)ix1 < (unsigned)Win);
        } else {
          // transposed conv, stride==2: out[o] += w[j] * x[(o+p-j)/2] when even
          int tz0 = oz0 + pad - jz, ty0 = oy0 + pad - jy, tx0 = ox0 + pad - jx;
          val0 = ((tz0 | ty0 | tx0) >= 0) && (((tz0 | ty0 | tx0) & 1) == 0);
          iz0 = tz0 >> 1; iy0 = ty0 >> 1; ix0 = tx0 >> 1;
          val0 = val0 && (iz0 < Din) && (iy0 < Hin) && (ix0 < Win);
          int tz1 = oz1 + pad - jz, ty1 = oy1 + pad - jy, tx1 = ox1 + pad - jx;
          val1 = ((tz1 | ty1 | tx1) >= 0) && (((tz1 | ty1 | tx1) & 1) == 0);
          iz1 = tz1 >> 1; iy1 = ty1 >> 1; ix1 = tx1 >> 1;
          val1 = val1 && (iz1 < Din) && (iy1 < Hin) && (ix1 < Win);
        }
        const long ib0 = (long)((iz0*Hin + iy0)*(long)Win + ix0) * CINP;
        const long ib1 = (long)((iz1*Hin + iy1)*(long)Win + ix1) * CINP;
        const int  cs  = (jy*KS + jx) * NCHK;

#pragma unroll
        for (int ci = 0; ci < NCHK; ++ci) {
          ABFrag a0, a1, b0, b1;
          const uint4 z4 = uint4{0u, 0u, 0u, 0u};
          const __bf16* ap0 = in + ib0 + ci*32 + khalf*8;
          const __bf16* ap1 = in + ib1 + ci*32 + khalf*8;
          a0.q[0] = val0 ? *(const uint4*)(ap0)      : z4;
          a0.q[1] = val0 ? *(const uint4*)(ap0 + 16) : z4;
          a1.q[0] = val1 ? *(const uint4*)(ap1)      : z4;
          a1.q[1] = val1 ? *(const uint4*)(ap1 + 16) : z4;
          // B from LDS: row (cs+ci), [ntp][kg][lane][32B]
          const unsigned char* bs = smem + (size_t)(cs + ci)*2048 + ((khalf*16) + lidx)*32;
          b0.q[0] = *(const uint4*)(bs);
          b0.q[1] = *(const uint4*)(bs + 16);
          b1.q[0] = *(const uint4*)(bs + 1024);
          b1.q[1] = *(const uint4*)(bs + 1040);
          acc00 = __builtin_amdgcn_wmma_f32_16x16x32_bf16(false, a0.v, false, b0.v, (short)0, acc00, false, false);
          acc01 = __builtin_amdgcn_wmma_f32_16x16x32_bf16(false, a0.v, false, b1.v, (short)0, acc01, false, false);
          acc10 = __builtin_amdgcn_wmma_f32_16x16x32_bf16(false, a1.v, false, b0.v, (short)0, acc10, false, false);
          acc11 = __builtin_amdgcn_wmma_f32_16x16x32_bf16(false, a1.v, false, b1.v, (short)0, acc11, false, false);
        }
      }
  }

  // Epilogue: C/D layout -> lane l holds N=l&15, VGPR v -> M = (l>>4)*8 + v.
  const int n0 = blockIdx.y * 32 + lidx;
#pragma unroll
  for (int v = 0; v < 8; ++v) {
    const int vox0 = mbase + khalf*8 + v;
    if (vox0 < Vout) {
      float msc = mask ? (float)mask[vox0] : 1.0f;
      out[(long)vox0 * Cout + n0]      = (__bf16)(acc00[v] * msc);
      out[(long)vox0 * Cout + n0 + 16] = (__bf16)(acc01[v] * msc);
    }
    const int vox1 = mbase + 16 + khalf*8 + v;
    if (vox1 < Vout) {
      float msc = mask ? (float)mask[vox1] : 1.0f;
      out[(long)vox1 * Cout + n0]      = (__bf16)(acc10[v] * msc);
      out[(long)vox1 * Cout + n0 + 16] = (__bf16)(acc11[v] * msc);
    }
  }
}

__global__ void gather_out(const __bf16* __restrict__ fin, const int* __restrict__ co,
                           float* __restrict__ out) {
  int i = blockIdx.x, c = threadIdx.x;
  if (i >= NPTS || c >= 32) return;
  int z = co[i*4+1], y = co[i*4+2], x = co[i*4+3];
  long vox = ((long)z*H0 + y)*W0 + x;
  out[(long)i*32 + c] = (float)fin[vox*32 + c];
}

// ---------------- driver ----------------

extern "C" void kernel_launch(void* const* d_in, const int* in_sizes, int n_in,
                              void* d_out, int out_size, void* d_ws, size_t ws_size,
                              hipStream_t stream) {
  (void)in_sizes; (void)n_in; (void)out_size; (void)ws_size;
  const float* feats   = (const float*)d_in[0];
  const int*   coors   = (const int*)d_in[1];
  const float* w_sub0  = (const float*)d_in[3];
  const float* w_sub1  = (const float*)d_in[4];
  const float* w_sub2  = (const float*)d_in[5];
  const float* w_sub3  = (const float*)d_in[6];
  const float* w_down0 = (const float*)d_in[7];
  const float* w_sub4  = (const float*)d_in[8];
  const float* w_sub5  = (const float*)d_in[9];
  const float* w_down1 = (const float*)d_in[10];
  const float* w_inv1  = (const float*)d_in[11];
  const float* w_inv0  = (const float*)d_in[12];

  char* ws = (char*)d_ws;
  size_t off = 0;
  auto alloc = [&](size_t bytes) -> char* {
    char* p = ws + off;
    off += (bytes + 1023) & ~(size_t)1023;   // keep TDM sources 1KB-aligned
    return p;
  };

  __bf16* A0  = (__bf16*)alloc((size_t)V0 * 64 * 2);   // full-res ping
  __bf16* B0  = (__bf16*)alloc((size_t)V0 * 64 * 2);   // full-res pong
  __bf16* m0  = (__bf16*)alloc((size_t)V0 * 2);        // occupancy mask
  __bf16* md  = (__bf16*)alloc((size_t)V1 * 2);        // downsampled mask
  __bf16* C1  = (__bf16*)alloc((size_t)V1 * 96 * 2);   // mid-res ping
  __bf16* Dm  = (__bf16*)alloc((size_t)V1 * 96 * 2);   // mid-res pong
  __bf16* E2  = (__bf16*)alloc((size_t)V2 * 96 * 2);   // low-res
  unsigned short* pw0  = (unsigned short*)alloc((size_t)27*32*32*2);
  unsigned short* pw1  = (unsigned short*)alloc((size_t)27*32*32*2);
  unsigned short* pw2  = (unsigned short*)alloc((size_t)27*32*64*2);
  unsigned short* pw3  = (unsigned short*)alloc((size_t)27*64*64*2);
  unsigned short* pwd0 = (unsigned short*)alloc((size_t)27*64*64*2);
  unsigned short* pw4  = (unsigned short*)alloc((size_t)27*64*96*2);
  unsigned short* pw5  = (unsigned short*)alloc((size_t)27*96*96*2);
  unsigned short* pwd1 = (unsigned short*)alloc((size_t)8*96*96*2);
  unsigned short* pwi1 = (unsigned short*)alloc((size_t)8*96*64*2);
  unsigned short* pwi0 = (unsigned short*)alloc((size_t)27*64*32*2);

  // Inputs: zero padded dense volume + mask, scatter points, build mid mask.
  {
    long nA = (long)V0 * 32;
    zero_bf16<<<dim3((unsigned)((nA + 255) / 256)), dim3(256), 0, stream>>>(A0, nA);
    zero_bf16<<<dim3((V0 + 255) / 256), dim3(256), 0, stream>>>(m0, (long)V0);
    scatter_pts<<<dim3((NPTS + 255) / 256), dim3(256), 0, stream>>>(feats, coors, A0, m0);
    mask_downsample<<<dim3((V1 + 255) / 256), dim3(256), 0, stream>>>(m0, md);
  }

  auto pack = [&](const float* w, unsigned short* wp, int taps, int Cin, int Cinp, int Cout) {
    int total = taps * Cinp * Cout;
    pack_weights<<<dim3((total + 255) / 256), dim3(256), 0, stream>>>(w, wp, taps, Cin, Cinp, Cout);
  };
  pack(w_sub0,  pw0,  27,  3, 32, 32);
  pack(w_sub1,  pw1,  27, 32, 32, 32);
  pack(w_sub2,  pw2,  27, 32, 32, 64);
  pack(w_sub3,  pw3,  27, 64, 64, 64);
  pack(w_down0, pwd0, 27, 64, 64, 64);
  pack(w_sub4,  pw4,  27, 64, 64, 96);
  pack(w_sub5,  pw5,  27, 96, 96, 96);
  pack(w_down1, pwd1,  8, 96, 96, 96);
  pack(w_inv1,  pwi1,  8, 96, 96, 64);
  pack(w_inv0,  pwi0, 27, 64, 64, 32);

#define LAUNCH_CONV(KS, CINP, TRANS, IN_, OUT_, WP_, MASK_, Di, Hi, Wi, Do_, Ho, Wo, COUT, S, P) \
  conv_wmma<KS, CINP, TRANS><<<dim3((unsigned)(((Do_)*(Ho)*(Wo) + 255) / 256), (COUT) / 32),      \
                               dim3(256), 0, stream>>>(                                           \
      IN_, OUT_, (const unsigned char*)(WP_), MASK_, Di, Hi, Wi, Do_, Ho, Wo, COUT, S, P)

  const __bf16* nomask = nullptr;
  // Full-res submanifold stack (masked by mask0)
  LAUNCH_CONV(3, 32, false, A0, B0, pw0,  m0,     D0, H0, W0, D0, H0, W0, 32, 1, 1);
  LAUNCH_CONV(3, 32, false, B0, A0, pw1,  m0,     D0, H0, W0, D0, H0, W0, 32, 1, 1);
  LAUNCH_CONV(3, 32, false, A0, B0, pw2,  m0,     D0, H0, W0, D0, H0, W0, 64, 1, 1);
  LAUNCH_CONV(3, 64, false, B0, A0, pw3,  m0,     D0, H0, W0, D0, H0, W0, 64, 1, 1);
  // Downsample to (14,64,64)
  LAUNCH_CONV(3, 64, false, A0, C1, pwd0, nomask, D0, H0, W0, D1, H1, W1, 64, 2, 1);
  LAUNCH_CONV(3, 64, false, C1, Dm, pw4,  md,     D1, H1, W1, D1, H1, W1, 96, 1, 1);
  LAUNCH_CONV(3, 96, false, Dm, C1, pw5,  md,     D1, H1, W1, D1, H1, W1, 96, 1, 1);
  // Downsample to (7,32,32)
  LAUNCH_CONV(2, 96, false, C1, E2, pwd1, nomask, D1, H1, W1, D2, H2, W2, 96, 2, 0);
  // Transposed convs back up
  LAUNCH_CONV(2, 96, true,  E2, Dm, pwi1, md,     D2, H2, W2, D1, H1, W1, 64, 2, 0);
  LAUNCH_CONV(3, 64, true,  Dm, B0, pwi0, m0,     D1, H1, W1, D0, H0, W0, 32, 2, 1);
#undef LAUNCH_CONV

  gather_out<<<dim3(NPTS), dim3(32), 0, stream>>>(B0, coors, (float*)d_out);
}
